// Conv2d_72052371357971
// MI455X (gfx1250) — matrix operand order repaired and checked by reading, not compiled, not run on hardware
//
#include <hip/hip_runtime.h>
#include <hip/hip_bf16.h>
#include <stdint.h>

// ---------------------------------------------------------------------------
// Fused conv-bank as one virtual GEMM:  Y[N,416] = X[N,768] * Wbig[768,416]
//   cols   0..149 : group1 (W1, 50 filters) at h-shifts 0,1,2
//   cols 150..249 : group2 (W2, 50 filters) at h-shifts 0,1
//   cols 250..405 : group3 (W3, 156 filters, full window)
//   cols 406..415 : zero pad
// Epilogue (per block, in LDS): maxpool over shifts + bias + concat -> out[N,256]
// Precision: X fp32 -> bf16 (RNE fptrunc) in-register, V_WMMA_F32_16X16X32_BF16,
// fp32 accumulate.
//
// Data movement:
//   A: per-lane coalesced global b128 loads (HBM stream) + vector cvt.
//   B: per-block double-buffered LDS staging (26KB/step, shared by all 8
//      waves) -> ds_load_b128 fragments, pipelined via sched_group_barrier.
// ---------------------------------------------------------------------------

typedef __attribute__((ext_vector_type(16))) __bf16 v16bf;
typedef __attribute__((ext_vector_type(16))) float  v16f;
typedef __attribute__((ext_vector_type(8)))  float  v8f;

#define KTOT    768
#define KSTEP   32
#define NSTEPS  24          // 768 / 32
#define CT      26          // column tiles of 16 -> 416 virtual cols
#define MT      64          // batch rows per block
#define THREADS 256
#define YSTRIDE 417         // padded LDS row stride
#define CHUNKB  (CT * 32 * 16 * 2)       // 26KB: one K-step of packed B
#define CHUNKQ  (CHUNKB / 16)            // 1664 uint4 per chunk
#define STG_IT  ((CHUNKQ + THREADS - 1) / THREADS)   // 7

#if defined(__has_builtin)
#if __has_builtin(__builtin_amdgcn_sched_group_barrier)
#define SGB(mask, n, id) __builtin_amdgcn_sched_group_barrier(mask, n, id)
#endif
#endif
#ifndef SGB
#define SGB(mask, n, id)
#endif

// ---------------------------------------------------------------------------
// Pack Wbig into per-lane WMMA B-fragment order (bf16):
//   Bp[ ((tile*NSTEPS + s)*32 + lane)*16 + j ]
// 16-bit B (32x16 tile): lanes 0-15 hold K=0..7,16..23 at N=lane; lanes 16-31 hold
// K=8..15,24..31 at N=lane-16; j indexes K within the lane.
// ---------------------------------------------------------------------------
__global__ void pack_w_kernel(const float* __restrict__ W1,
                              const float* __restrict__ W2,
                              const float* __restrict__ W3,
                              unsigned short* __restrict__ Bp) {
  int idx = blockIdx.x * blockDim.x + threadIdx.x;
  const int total = CT * NSTEPS * 32 * 16;
  if (idx >= total) return;

  int j    = idx & 15;
  int lane = (idx >> 4) & 31;
  int ts   = idx >> 9;           // tile*NSTEPS + s
  int s    = ts % NSTEPS;
  int t    = ts / NSTEPS;

  int col  = t * 16 + (lane & 15);
  int k    = s * 32 + ((lane & 16) >> 1) + (j & 7) + ((j & 8) << 1);      // lane<16 -> K 0..7,16..23, else 8..15,24..31

  float v = 0.0f;
  if (col < 150) {                           // group 1: W1 at shift h
    int h = col / 50, i = col % 50;
    int kk = k - h * 256;
    if (kk >= 0 && kk < 256) v = W1[i * 256 + kk];
  } else if (col < 250) {                    // group 2: W2 spans 512 rows at shift
    int blk = (col - 150) / 50, i = (col - 150) % 50;
    int kk = k - blk * 256;
    if (kk >= 0 && kk < 512) v = W2[(i * 2 + (kk >> 8)) * 256 + (kk & 255)];
  } else if (col < 406) {                    // group 3: full 768-row filter
    int i = col - 250;
    v = W3[i * 768 + k];
  }
  Bp[idx] = __builtin_bit_cast(unsigned short, (__bf16)v);   // RNE fptrunc
}

// ---------------------------------------------------------------------------
// Main GEMM + fused epilogue.
// Block: 256 threads = 8 wave32. wave w: row-tile rt = w>>1 (4 x 16 rows),
// column half ch = w&1 (13 tiles each).
// Dynamic LDS: [ B chunk buf0 | B chunk buf1 ] aliased over the front of the
// 64 x 417 fp32 epilogue Y strip (used only after the final loop barrier).
// ---------------------------------------------------------------------------
__global__ __launch_bounds__(THREADS) void conv_gemm_kernel(
    const float* __restrict__ X,
    const unsigned short* __restrict__ Bp,
    const float* __restrict__ b1,
    const float* __restrict__ b2,
    const float* __restrict__ b3,
    float* __restrict__ out) {
  extern __shared__ char smem_raw[];
  float* Ysm = (float*)smem_raw;            // [MT][YSTRIDE] (epilogue)

  const int tid  = threadIdx.x;
  const int lane = tid & 31;
  const int wave = tid >> 5;
  const int rt   = wave >> 1;              // 0..3
  const int ch   = wave & 1;               // 0..1
  const long row0 = (long)blockIdx.x * MT;

  v8f acc[13];
  const v8f vzero = {0.f, 0.f, 0.f, 0.f, 0.f, 0.f, 0.f, 0.f};
#pragma unroll
  for (int t = 0; t < 13; ++t) acc[t] = vzero;

  // A fragment geometry (16-bit A 16x32): lanes 0-15 hold K 0..7 & 16..23,
  // lanes 16-31 hold K 8..15 & 24..31, at M = lane&15.
  const float* Xa =
      X + (row0 + 16 * rt + (lane & 15)) * KTOT + ((lane & 16) >> 1);

  const uint4* Bq = (const uint4*)Bp;   // global packed B, uint4 granules

  // --- prologue: stage step-0 B chunk into LDS buffer 0
  {
    uint4* Lq = (uint4*)smem_raw;
#pragma unroll
    for (int i = 0; i < STG_IT; ++i) {
      int q = tid + i * THREADS;
      if (q < CHUNKQ) {
        int t = q >> 6, r4 = q & 63;            // 64 uint4 per tile-chunk
        Lq[q] = Bq[t * (NSTEPS * 64) + r4];     // s = 0
      }
    }
  }
  __syncthreads();

  for (int s = 0; s < NSTEPS; ++s) {
    // ---- issue staging global loads for step s+1 (in flight under WMMAs)
    uint4 stg[STG_IT];
    if (s + 1 < NSTEPS) {
#pragma unroll
      for (int i = 0; i < STG_IT; ++i) {
        int q = tid + i * THREADS;
        if (q < CHUNKQ) {
          int t = q >> 6, r4 = q & 63;
          stg[i] = Bq[t * (NSTEPS * 64) + (s + 1) * 64 + r4];
        }
      }
    }

    // ---- A fragment: two 8-float runs (k, k+16), 4 x global b128, cvt bf16
    const float* xp = Xa + s * KSTEP;
    float4 a0 = *(const float4*)(xp + 0);
    float4 a1 = *(const float4*)(xp + 4);
    float4 a2 = *(const float4*)(xp + 16);
    float4 a3 = *(const float4*)(xp + 20);
    if (s + 2 < NSTEPS) __builtin_prefetch(Xa + (s + 2) * KSTEP, 0, 1);

    v16f af;
    af[0]  = a0.x; af[1]  = a0.y; af[2]  = a0.z; af[3]  = a0.w;
    af[4]  = a1.x; af[5]  = a1.y; af[6]  = a1.z; af[7]  = a1.w;
    af[8]  = a2.x; af[9]  = a2.y; af[10] = a2.z; af[11] = a2.w;
    af[12] = a3.x; af[13] = a3.y; af[14] = a3.z; af[15] = a3.w;
    v16bf av = __builtin_convertvector(af, v16bf);   // RNE fptrunc to bf16

    // ---- consume B fragments from LDS buffer (s&1), 2-deep pipelined
    const char* buf = smem_raw + (s & 1) * CHUNKB;
    const int  fbase = ((ch * 13) << 10) | (lane << 5);
    union F { uint4 q[2]; v16bf v; };
    F bcur, bnxt;
    {
      const uint4* fp = (const uint4*)(buf + fbase);
      bcur.q[0] = fp[0];
      bcur.q[1] = fp[1];
    }
#pragma unroll
    for (int t = 0; t < 13; ++t) {
      if (t + 1 < 13) {
        const uint4* fp = (const uint4*)(buf + fbase + ((t + 1) << 10));
        bnxt.q[0] = fp[0];
        bnxt.q[1] = fp[1];
      }
      acc[t] = __builtin_amdgcn_wmma_f32_16x16x32_bf16(
          false, av, false, bcur.v, (short)0, acc[t], false, false);
      bcur = bnxt;
    }

    // Pin the pipeline: ds-reads (0x100) for tile t+1 stay ahead of WMMA
    // (MFMA/WMMA group, 0x8) for tile t.  26 ds-reads, 13 WMMAs per step.
    SGB(0x100, 4, 0);                       // tiles 0,1 fragments
#pragma unroll
    for (int t = 0; t < 11; ++t) {
      SGB(0x008, 1, 0);                     // WMMA t
      SGB(0x100, 2, 0);                     // fragments for tile t+2
    }
    SGB(0x008, 1, 0);                       // WMMA 11
    SGB(0x008, 1, 0);                       // WMMA 12

    // ---- commit staged chunk to the other LDS buffer, then step barrier
    if (s + 1 < NSTEPS) {
      uint4* Lq = (uint4*)(smem_raw + ((s + 1) & 1) * CHUNKB);
#pragma unroll
      for (int i = 0; i < STG_IT; ++i) {
        int q = tid + i * THREADS;
        if (q < CHUNKQ) Lq[q] = stg[i];
      }
    }
    __syncthreads();
  }

  // Scatter accumulators to LDS Y strip.
  // C/D layout: VGPR v, lanes 0-15 -> (M=v, N=lane); lanes 16-31 -> (M=v+8).
  {
    const int nloc = lane & 15;
    const int mofs = 16 * rt + ((lane >> 4) ? 8 : 0);
#pragma unroll
    for (int t = 0; t < 13; ++t) {
      const int col = (ch * 13 + t) * 16 + nloc;
#pragma unroll
      for (int v = 0; v < 8; ++v) {
        Ysm[(mofs + v) * YSTRIDE + col] = acc[t][v];
      }
    }
  }
  __syncthreads();

  // Fused epilogue: maxpool over shifts + bias + concat -> 256 channels/row.
  for (int idx = tid; idx < MT * 256; idx += THREADS) {
    const int m = idx >> 8;
    const int j = idx & 255;
    float r;
    if (j < 50) {
      float v0 = Ysm[m * YSTRIDE + j]       + b1[j * 3 + 0];
      float v1 = Ysm[m * YSTRIDE + j + 50]  + b1[j * 3 + 1];
      float v2 = Ysm[m * YSTRIDE + j + 100] + b1[j * 3 + 2];
      r = fmaxf(v0, fmaxf(v1, v2));
    } else if (j < 100) {
      const int i = j - 50;
      float v0 = Ysm[m * YSTRIDE + 150 + i] + b2[i * 2 + 0];
      float v1 = Ysm[m * YSTRIDE + 200 + i] + b2[i * 2 + 1];
      r = fmaxf(v0, v1);
    } else {
      const int i = j - 100;
      r = Ysm[m * YSTRIDE + 250 + i] + b3[i];
    }
    out[(row0 + m) * 256 + j] = r;
  }
}

extern "C" void kernel_launch(void* const* d_in, const int* in_sizes, int n_in,
                              void* d_out, int out_size, void* d_ws, size_t ws_size,
                              hipStream_t stream) {
  const float* x  = (const float*)d_in[0];   // [65536,3,256,1]
  const float* W1 = (const float*)d_in[1];   // [50,256]
  const float* b1 = (const float*)d_in[2];   // [50,3]
  const float* W2 = (const float*)d_in[3];   // [50,2,256]
  const float* b2 = (const float*)d_in[4];   // [50,2]
  const float* W3 = (const float*)d_in[5];   // [156,3,256]
  const float* b3 = (const float*)d_in[6];   // [156]
  float* out = (float*)d_out;

  unsigned short* Bp = (unsigned short*)d_ws;     // 26*24*32*16*2 = 638,976 B

  const int packTotal = CT * NSTEPS * 32 * 16;
  pack_w_kernel<<<(packTotal + 255) / 256, 256, 0, stream>>>(W1, W2, W3, Bp);

  const int nRows = 65536;
  const size_t lds = (size_t)MT * YSTRIDE * sizeof(float);   // 106,752 B
  conv_gemm_kernel<<<nRows / MT, THREADS, lds, stream>>>(x, Bp, b1, b2, b3, out);
}
